// LGSA_30279519436939
// MI455X (gfx1250) — matrix operand order repaired and checked by reading, not compiled, not run on hardware
//
#include <hip/hip_runtime.h>
#include <hip/hip_bf16.h>
#include <math.h>

typedef __attribute__((ext_vector_type(16))) _Float16 v16h;
typedef __attribute__((ext_vector_type(8)))  _Float16 v8h;
typedef __attribute__((ext_vector_type(8)))  float    v8f;

#define USE_ASYNC_LDS 1

// ---------------- problem constants ----------------
static const int BB   = 32;
static const int CC   = 384;
static const int HH   = 32;
static const int WWd  = 32;
static const int NT   = 1024;          // tokens
static const int NHh  = 6;
static const int DHh  = 64;
static const int HID2 = 2304;

// ---------------- device helpers ----------------
__device__ __forceinline__ float wred_sum(float v) {
#pragma unroll
  for (int o = 16; o > 0; o >>= 1) v += __shfl_xor(v, o, 32);
  return v;
}
__device__ __forceinline__ float wred_max(float v) {
#pragma unroll
  for (int o = 16; o > 0; o >>= 1) v = fmaxf(v, __shfl_xor(v, o, 32));
  return v;
}
__device__ __forceinline__ v16h cat8(v8h lo, v8h hi) {
  return __builtin_shufflevector(lo, hi, 0,1,2,3,4,5,6,7,8,9,10,11,12,13,14,15);
}

__device__ __forceinline__ void async_b128_to_lds(const _Float16* gsrc, _Float16* lds) {
#if USE_ASYNC_LDS
  unsigned laddr = (unsigned)(size_t)lds;   // low 32 bits of flat LDS pointer = LDS byte addr
  asm volatile("global_load_async_to_lds_b128 %0, %1, off"
               :: "v"(laddr), "v"(gsrc) : "memory");
#else
  *(v8h*)lds = *(const v8h*)gsrc;
#endif
}
__device__ __forceinline__ void async_wait0() {
#if USE_ASYNC_LDS
#if __has_builtin(__builtin_amdgcn_s_wait_asynccnt)
  __builtin_amdgcn_s_wait_asynccnt(0);
#else
  asm volatile("s_wait_asynccnt 0x0" ::: "memory");
#endif
#endif
}

// ---------------- tiny prep kernels ----------------
__global__ void k_four(float* __restrict__ fy, float* __restrict__ fx) {
  int t = blockIdx.x * blockDim.x + threadIdx.x;
  if (t >= 2048) return;
  int which = t >> 10, idx = (t >> 5) & 31, j = t & 31;
  const float scale = 6.283185307179586f;
  float e = (float)(idx + 1) / (32.0f + 1e-6f) * scale;
  int tp = j >> 1;
  float dt = powf(10000.0f, (float)(2 * tp) / 32.0f);
  float p = e / dt;
  float val = (j & 1) ? cosf(p) : sinf(p);
  (which ? fx : fy)[idx * 32 + j] = val;
}

__global__ void k_pos(const float* __restrict__ fy, const float* __restrict__ fx,
                      const float* __restrict__ pw, const float* __restrict__ pb,
                      float* __restrict__ pos) {
  int t = blockIdx.x * blockDim.x + threadIdx.x;
  if (t >= NT * CC) return;
  int c = t % CC, n = t / CC;
  int hh = n >> 5, ww = n & 31;
  const float* wrow = pw + (size_t)c * 64;
  float s = pb[c];
#pragma unroll
  for (int j = 0; j < 32; j++) s += fy[hh * 32 + j] * wrow[j];
#pragma unroll
  for (int j = 0; j < 32; j++) s += fx[ww * 32 + j] * wrow[32 + j];
  pos[t] = s;   // (n, c)
}

__global__ void k_cvt(const float* __restrict__ s, _Float16* __restrict__ d, int n) {
  int t = blockIdx.x * blockDim.x + threadIdx.x;
  if (t < n) d[t] = (_Float16)s[t];
}

// LayerNorm over C; input layout (B,C,N), outputs layout (B,N,C). One wave per token.
__global__ void k_ln(const float* __restrict__ xin, const float* __restrict__ pos,
                     const float* __restrict__ w, const float* __restrict__ bb,
                     float* __restrict__ outF, _Float16* __restrict__ outH) {
  int lane = threadIdx.x & 31;
  int tok = blockIdx.x * (blockDim.x >> 5) + (threadIdx.x >> 5);
  if (tok >= BB * NT) return;
  int b = tok >> 10, n = tok & 1023;
  const float* base = xin + (size_t)b * CC * NT + n;
  float v[12]; float s = 0.f;
#pragma unroll
  for (int i = 0; i < 12; i++) {
    int c = lane + 32 * i;
    float t = base[(size_t)c * NT];
    if (pos) t += pos[n * CC + c];
    v[i] = t; s += t;
  }
  s = wred_sum(s);
  float mean = s * (1.0f / CC);
  float vs = 0.f;
#pragma unroll
  for (int i = 0; i < 12; i++) { float d = v[i] - mean; vs += d * d; }
  vs = wred_sum(vs);
  float inv = rsqrtf(vs * (1.0f / CC) + 1e-6f);
  size_t ob = (size_t)tok * CC;
#pragma unroll
  for (int i = 0; i < 12; i++) {
    int c = lane + 32 * i;
    float o = (v[i] - mean) * inv * w[c] + bb[c];
    if (outF) outF[ob + c] = o;
    if (outH) outH[ob + c] = (_Float16)o;
  }
}

// ---------------- WMMA GEMM epilogue modes ----------------
#define EPI_F32  0
#define EPI_F16  1
#define EPI_GELU 2
#define EPI_RES  3

__device__ __forceinline__ void gemm_epilogue(
    v8f acc0, v8f acc1, v8f acc2, v8f acc3,
    int m0, int n0, int lane, long long obase,
    const float* __restrict__ bias,
    float* __restrict__ outF, _Float16* __restrict__ outH,
    long long oSM, long long oSN,
    const float* __restrict__ resid, const float* __restrict__ gamma, int epi) {
  int mhi = (lane < 16) ? 0 : 8;
#pragma unroll
  for (int t = 0; t < 4; t++) {
    v8f acc = (t == 0) ? acc0 : (t == 1) ? acc1 : (t == 2) ? acc2 : acc3;
    int ncol = n0 + t * 16 + (lane & 15);
    float bv = bias ? bias[ncol] : 0.f;
#pragma unroll
    for (int rr = 0; rr < 8; rr++) {
      int mm = m0 + rr + mhi;
      float val = acc[rr] + bv;
      long long idx = obase + (long long)mm * oSM + (long long)ncol * oSN;
      if (epi == EPI_F32) {
        outF[idx] = val;
      } else if (epi == EPI_F16) {
        outH[idx] = (_Float16)val;
      } else if (epi == EPI_GELU) {
        float ge = 0.5f * val * (1.f + erff(val * 0.70710678118f));
        outH[idx] = (_Float16)ge;
      } else {
        outF[idx] = resid[idx] + val * gamma[ncol];
      }
    }
  }
}

// ---------------- direct-global WMMA GEMM (small batched attention GEMMs) ----------------
// D = A(MxK, row-major, lda) * B (element (k,n) at B + n*sBN + k) + bias[n]
// Wave tile: 16(M) x 64(N). Two-level batching (outer, inner) with strides.
__global__ void k_gemm(int M, int N, int K, int bO, int bI,
                       const _Float16* __restrict__ A, int lda, long long sAo, long long sAi,
                       const _Float16* __restrict__ Bm, int sBN, long long sBo, long long sBi,
                       const float* __restrict__ bias,
                       float* __restrict__ outF, _Float16* __restrict__ outH,
                       long long oSM, long long oSN, long long sOo, long long sOi,
                       const float* __restrict__ resid, const float* __restrict__ gamma,
                       int epi) {
  int lane = threadIdx.x & 31;
  int wave = blockIdx.x * (blockDim.x >> 5) + (threadIdx.x >> 5);
  int mt = M >> 4, nt = N >> 6;
  int tpb = mt * nt;
  long long total = (long long)bO * bI * tpb;
  if (wave >= total) return;
  int g = wave / tpb, r = wave % tpb;
  int go = g / bI, gi = g % bI;
  int i = r / nt, j = r % nt;
  int m0 = i << 4, n0 = j << 6;

  const _Float16* Ab = A + (size_t)go * sAo + (size_t)gi * sAi;
  const _Float16* Bb = Bm + (size_t)go * sBo + (size_t)gi * sBi;

  const _Float16* arow = Ab + (size_t)(m0 + (lane & 15)) * lda + ((lane >> 4) << 4);
  const _Float16* bc0 = Bb + (size_t)(n0 + (lane & 15)) * sBN + ((lane >> 4) << 4);
  const _Float16* bc1 = bc0 + (size_t)16 * sBN;
  const _Float16* bc2 = bc0 + (size_t)32 * sBN;
  const _Float16* bc3 = bc0 + (size_t)48 * sBN;

  v8f acc0 = {}, acc1 = {}, acc2 = {}, acc3 = {};
  for (int k0 = 0; k0 < K; k0 += 32) {
    v16h af = cat8(*(const v8h*)(arow + k0), *(const v8h*)(arow + k0 + 8));
    v16h b0 = cat8(*(const v8h*)(bc0 + k0), *(const v8h*)(bc0 + k0 + 8));
    v16h b1 = cat8(*(const v8h*)(bc1 + k0), *(const v8h*)(bc1 + k0 + 8));
    v16h b2 = cat8(*(const v8h*)(bc2 + k0), *(const v8h*)(bc2 + k0 + 8));
    v16h b3 = cat8(*(const v8h*)(bc3 + k0), *(const v8h*)(bc3 + k0 + 8));
    acc0 = __builtin_amdgcn_wmma_f32_16x16x32_f16(false, af, false, b0, (short)0, acc0, false, false);
    acc1 = __builtin_amdgcn_wmma_f32_16x16x32_f16(false, af, false, b1, (short)0, acc1, false, false);
    acc2 = __builtin_amdgcn_wmma_f32_16x16x32_f16(false, af, false, b2, (short)0, acc2, false, false);
    acc3 = __builtin_amdgcn_wmma_f32_16x16x32_f16(false, af, false, b3, (short)0, acc3, false, false);
  }
  long long obase = (long long)go * sOo + (long long)gi * sOi;
  gemm_epilogue(acc0, acc1, acc2, acc3, m0, n0, lane, obase,
                bias, outF, outH, oSM, oSN, resid, gamma, epi);
}

// ---------------- LDS-staged WMMA GEMM (big GEMMs; M % 128 == 0) ----------------
// Block = 8 waves covering 128(M) x 64(N); B tile (64n x 32k) staged in LDS via
// async global->LDS copies, double-buffered, shared by all 8 waves.
__global__ void k_gemm_lds(int M, int N, int K, int bO, int bI,
                           const _Float16* __restrict__ A, int lda, long long sAo, long long sAi,
                           const _Float16* __restrict__ Bm, int sBN, long long sBo, long long sBi,
                           const float* __restrict__ bias,
                           float* __restrict__ outF, _Float16* __restrict__ outH,
                           long long oSM, long long oSN, long long sOo, long long sOi,
                           const float* __restrict__ resid, const float* __restrict__ gamma,
                           int epi) {
  __shared__ _Float16 bs[2][64 * 32];    // 2 x 4KB
  int tid = threadIdx.x;
  int lane = tid & 31, wv = tid >> 5;
  int mt = M >> 7, nt = N >> 6;
  int tpb = mt * nt;
  long long blk = blockIdx.x;
  int g = (int)(blk / tpb), r = (int)(blk % tpb);
  int go = g / bI, gi = g % bI;
  int i = r / nt, j = r % nt;
  int m0 = (i << 7) + (wv << 4), n0 = j << 6;

  const _Float16* Ab = A + (size_t)go * sAo + (size_t)gi * sAi;
  const _Float16* Bb = Bm + (size_t)go * sBo + (size_t)gi * sBi;

  // staging assignment: thread -> one b128 chunk of the 64x32 B tile
  int srow = tid >> 2, sseg = (tid & 3) << 3;          // row 0..63, seg offset in halves
  const _Float16* gstage = Bb + (size_t)(n0 + srow) * sBN + sseg;
  _Float16* lstage0 = &bs[0][srow * 32 + sseg];
  _Float16* lstage1 = &bs[1][srow * 32 + sseg];

  const _Float16* arow = Ab + (size_t)(m0 + (lane & 15)) * lda + ((lane >> 4) << 4);
  // per-wave LDS read base: lane holds column (lane&15) within each 16-col group
  const int lb_off = ((lane >> 4) << 4);               // khalf*16

  int nsteps = K >> 5;
  // prologue: stage step 0 into buffer 0
  async_b128_to_lds(gstage, lstage0);
  async_wait0();
  __syncthreads();

  v8f acc0 = {}, acc1 = {}, acc2 = {}, acc3 = {};
  for (int s = 0; s < nsteps; s++) {
    int cur = s & 1;
    if (s + 1 < nsteps)
      async_b128_to_lds(gstage + ((s + 1) << 5), cur ? lstage0 : lstage1);

    int k0 = s << 5;
    v16h af = cat8(*(const v8h*)(arow + k0), *(const v8h*)(arow + k0 + 8));
    const _Float16* lb = &bs[cur][(lane & 15) * 32 + lb_off];
    v16h b0 = cat8(*(const v8h*)(lb),            *(const v8h*)(lb + 8));
    v16h b1 = cat8(*(const v8h*)(lb + 16 * 32),  *(const v8h*)(lb + 16 * 32 + 8));
    v16h b2 = cat8(*(const v8h*)(lb + 32 * 32),  *(const v8h*)(lb + 32 * 32 + 8));
    v16h b3 = cat8(*(const v8h*)(lb + 48 * 32),  *(const v8h*)(lb + 48 * 32 + 8));
    acc0 = __builtin_amdgcn_wmma_f32_16x16x32_f16(false, af, false, b0, (short)0, acc0, false, false);
    acc1 = __builtin_amdgcn_wmma_f32_16x16x32_f16(false, af, false, b1, (short)0, acc1, false, false);
    acc2 = __builtin_amdgcn_wmma_f32_16x16x32_f16(false, af, false, b2, (short)0, acc2, false, false);
    acc3 = __builtin_amdgcn_wmma_f32_16x16x32_f16(false, af, false, b3, (short)0, acc3, false, false);

    async_wait0();
    __syncthreads();
  }

  long long obase = (long long)go * sOo + (long long)gi * sOi;
  gemm_epilogue(acc0, acc1, acc2, acc3, m0, n0, lane, obase,
                bias, outF, outH, oSM, oSN, resid, gamma, epi);
}

// L2-normalize over tokens for each (b,c); write transposed (B,C,N) f16.
__global__ void k_l2n(const float* __restrict__ src, _Float16* __restrict__ dst) {
  int lane = threadIdx.x & 31;
  int wave = blockIdx.x * (blockDim.x >> 5) + (threadIdx.x >> 5);
  if (wave >= BB * CC) return;
  int b = wave / CC, c = wave % CC;
  const float* p = src + (size_t)b * NT * CC + c;
  float v[32]; float s = 0.f;
#pragma unroll
  for (int i = 0; i < 32; i++) { v[i] = p[(size_t)(lane + 32 * i) * CC]; s += v[i] * v[i]; }
  s = wred_sum(s);
  float sc = 1.f / fmaxf(sqrtf(s), 1e-12f);
  _Float16* q = dst + ((size_t)b * CC + c) * NT;
#pragma unroll
  for (int i = 0; i < 32; i++) q[lane + 32 * i] = (_Float16)(v[i] * sc);
}

// dwconv(xn) -> BN -> SiLU, gate vc -> v_h (B,N,C) f16
__global__ void k_dwgate(const float* __restrict__ xn, const float* __restrict__ vc,
                         const float* __restrict__ dww, const float* __restrict__ dwb,
                         const float* __restrict__ bg, const float* __restrict__ bb2,
                         const float* __restrict__ bm, const float* __restrict__ bv,
                         _Float16* __restrict__ vh) {
  size_t t = (size_t)blockIdx.x * blockDim.x + threadIdx.x;
  if (t >= (size_t)BB * NT * CC) return;
  int c = (int)(t % CC); size_t r = t / CC;
  int n = (int)(r & 1023); int b = (int)(r >> 10);
  int hh = n >> 5, ww = n & 31;
  float acc = dwb[c];
  const float* img = xn + (size_t)b * NT * CC;
  const float* wk = dww + c * 9;
#pragma unroll
  for (int dy = 0; dy < 3; dy++) {
    int y = hh + dy - 1; if (y < 0 || y > 31) continue;
#pragma unroll
    for (int dx = 0; dx < 3; dx++) {
      int xw = ww + dx - 1; if (xw < 0 || xw > 31) continue;
      acc += wk[dy * 3 + dx] * img[(size_t)(y * 32 + xw) * CC + c];
    }
  }
  float g = (acc - bm[c]) * rsqrtf(bv[c] + 1e-5f) * bg[c] + bb2[c];
  float sg = g / (1.f + expf(-g));          // SiLU
  vh[t] = (_Float16)(vc[t] * sg);
}

// softmax rows of attn (B,h,d,d) with temp[h]; write f16
__global__ void k_smax(const float* __restrict__ attnF, const float* __restrict__ temp,
                       _Float16* __restrict__ attnH) {
  int lane = threadIdx.x & 31;
  int row = blockIdx.x * (blockDim.x >> 5) + (threadIdx.x >> 5);
  if (row >= BB * NHh * DHh) return;
  int h = (row >> 6) % NHh;
  float t = temp[h];
  const float* p = attnF + (size_t)row * 64;
  float v0 = p[lane] * t, v1 = p[lane + 32] * t;
  float m = wred_max(fmaxf(v0, v1));
  float e0 = expf(v0 - m), e1 = expf(v1 - m);
  float inv = 1.f / wred_sum(e0 + e1);
  _Float16* q = attnH + (size_t)row * 64;
  q[lane] = (_Float16)(e0 * inv);
  q[lane + 32] = (_Float16)(e1 * inv);
}

// residual 1 with the faithful swapaxes/reshape scramble
__global__ void k_resid1(const float* __restrict__ x, const float* __restrict__ projf,
                         const float* __restrict__ gamma1, float* __restrict__ x1) {
  size_t t = (size_t)blockIdx.x * blockDim.x + threadIdx.x;
  if (t >= (size_t)BB * CC * NT) return;
  int i = (int)(t % (CC * NT)); int b = (int)(t / (CC * NT));
  int hh = i / (CC * WWd);            // / 12288
  int rem = i % (CC * WWd);
  int cc = rem / WWd;
  int ww = rem % WWd;
  float xa = gamma1[cc] * projf[((size_t)b * NT + (size_t)hh * WWd + ww) * CC + cc];
  x1[t] = x[t] + xa;
}

// MLP depthwise conv + exact GELU ; h1 (B,N,hid) f16 -> h2 (B,N,hid) f16
__global__ void k_dw2(const _Float16* __restrict__ h1, const float* __restrict__ mdw,
                      const float* __restrict__ mdb, _Float16* __restrict__ h2) {
  size_t t = (size_t)blockIdx.x * blockDim.x + threadIdx.x;
  if (t >= (size_t)BB * NT * HID2) return;
  int o = (int)(t % HID2); size_t r = t / HID2;
  int n = (int)(r & 1023); int b = (int)(r >> 10);
  int hh = n >> 5, ww = n & 31;
  float acc = mdb[o];
  const _Float16* img = h1 + (size_t)b * NT * HID2;
  const float* wk = mdw + o * 9;
#pragma unroll
  for (int dy = 0; dy < 3; dy++) {
    int y = hh + dy - 1; if (y < 0 || y > 31) continue;
#pragma unroll
    for (int dx = 0; dx < 3; dx++) {
      int xw = ww + dx - 1; if (xw < 0 || xw > 31) continue;
      acc += wk[dy * 3 + dx] * (float)img[(size_t)(y * 32 + xw) * HID2 + o];
    }
  }
  float ge = 0.5f * acc * (1.f + erff(acc * 0.70710678118f));
  h2[t] = (_Float16)ge;
}

// ---------------- host side ----------------
static inline int gemm_blocks(int M, int N, int bO, int bI) {
  long long waves = (long long)bO * bI * (M / 16) * (N / 64);
  return (int)((waves + 7) / 8);     // 8 waves (256 threads) per block
}
static inline int gemm_blocks_lds(int M, int N, int bO, int bI) {
  return (int)((long long)bO * bI * (M / 128) * (N / 64));
}
static inline int cdiv(long long a, int b) { return (int)((a + b - 1) / b); }

extern "C" void kernel_launch(void* const* d_in, const int* in_sizes, int n_in,
                              void* d_out, int out_size, void* d_ws, size_t ws_size,
                              hipStream_t stream) {
  const float* x      = (const float*)d_in[0];
  const float* pos_w  = (const float*)d_in[1];
  const float* pos_b  = (const float*)d_in[2];
  const float* ln1_w  = (const float*)d_in[3];
  const float* ln1_b  = (const float*)d_in[4];
  const float* q_w    = (const float*)d_in[5];
  const float* q_b    = (const float*)d_in[6];
  const float* k_w    = (const float*)d_in[7];
  const float* k_b    = (const float*)d_in[8];
  const float* vc_w   = (const float*)d_in[9];
  const float* vc_b   = (const float*)d_in[10];
  const float* dw_w   = (const float*)d_in[11];
  const float* dw_b   = (const float*)d_in[12];
  const float* bn_g   = (const float*)d_in[13];
  const float* bn_b   = (const float*)d_in[14];
  const float* bn_m   = (const float*)d_in[15];
  const float* bn_v   = (const float*)d_in[16];
  const float* temp   = (const float*)d_in[17];
  const float* proj_w = (const float*)d_in[18];
  const float* proj_b = (const float*)d_in[19];
  const float* gamma1 = (const float*)d_in[20];
  const float* ln2_w  = (const float*)d_in[21];
  const float* ln2_b  = (const float*)d_in[22];
  const float* mlp_w1 = (const float*)d_in[23];
  const float* mlp_b1 = (const float*)d_in[24];
  const float* mlp_dw = (const float*)d_in[25];
  const float* mlp_db = (const float*)d_in[26];
  const float* mlp_w2 = (const float*)d_in[27];
  const float* mlp_b2 = (const float*)d_in[28];
  const float* gamma2 = (const float*)d_in[29];
  float* out = (float*)d_out;

  char* wsb = (char*)d_ws;
  // ---- persistent region ----
  float*    fy    = (float*)(wsb + 0);
  float*    fx    = (float*)(wsb + 4096);
  float*    pos   = (float*)(wsb + 8192);                 // 1.5 MB
  _Float16* wq_h  = (_Float16*)(wsb + 1581056);
  _Float16* wk_h  = (_Float16*)(wsb + 1875968);
  _Float16* wv_h  = (_Float16*)(wsb + 2170880);
  _Float16* wp_h  = (_Float16*)(wsb + 2465792);
  _Float16* w1_h  = (_Float16*)(wsb + 2760704);
  _Float16* w2_h  = (_Float16*)(wsb + 4530176);
  float*    attnF = (float*)(wsb + 6299648);
  _Float16* attnH = (_Float16*)(wsb + 9445376);
  // ---- arena (48 MiB slots), with reuse ----
  const size_t AR = 16777216, SLOT = 50331648;
  float*    xn_f  = (float*)   (wsb + AR + 0 * SLOT);
  _Float16* xn_h  = (_Float16*)(wsb + AR + 1 * SLOT);
  float*    q_f   = (float*)   (wsb + AR + 2 * SLOT);
  float*    k_f   = (float*)   (wsb + AR + 3 * SLOT);
  float*    vc_f  = (float*)   (wsb + AR + 4 * SLOT);
  _Float16* qt_h  = (_Float16*)(wsb + AR + 5 * SLOT);
  _Float16* kt_h  = (_Float16*)(wsb + AR + 5 * SLOT + 25165824);
  _Float16* v_h   = (_Float16*)(wsb + AR + 2 * SLOT);   // reuse q_f
  _Float16* xo_h  = (_Float16*)(wsb + AR + 3 * SLOT);   // reuse k_f
  float*    projf = (float*)   (wsb + AR + 4 * SLOT);   // reuse vc_f
  float*    x1    = (float*)   (wsb + AR + 0 * SLOT);   // reuse xn_f
  _Float16* xn2_h = (_Float16*)(wsb + AR + 1 * SLOT);   // reuse xn_h
  _Float16* h1_h  = (_Float16*)(wsb + AR + 2 * SLOT);   // spans slots 2..4
  _Float16* h2_h  = (_Float16*)(wsb + AR + 5 * SLOT);   // spans slots 5..7

  const long long BNC  = (long long)BB * NT * CC;   // 12.6M
  const long long BNH  = (long long)BB * NT * HID2; // 75.5M

  // 1) positional encoding
  k_four<<<8, 256, 0, stream>>>(fy, fx);
  k_pos<<<cdiv(NT * CC, 256), 256, 0, stream>>>(fy, fx, pos_w, pos_b, pos);

  // 2) weight conversions to f16
  k_cvt<<<cdiv(CC * CC, 256), 256, 0, stream>>>(q_w, wq_h, CC * CC);
  k_cvt<<<cdiv(CC * CC, 256), 256, 0, stream>>>(k_w, wk_h, CC * CC);
  k_cvt<<<cdiv(CC * CC, 256), 256, 0, stream>>>(vc_w, wv_h, CC * CC);
  k_cvt<<<cdiv(CC * CC, 256), 256, 0, stream>>>(proj_w, wp_h, CC * CC);
  k_cvt<<<cdiv(HID2 * CC, 256), 256, 0, stream>>>(mlp_w1, w1_h, HID2 * CC);
  k_cvt<<<cdiv(HID2 * CC, 256), 256, 0, stream>>>(mlp_w2, w2_h, HID2 * CC);

  // 3) LN1 (adds pos)
  k_ln<<<BB * NT / 8, 256, 0, stream>>>(x, pos, ln1_w, ln1_b, xn_f, xn_h);

  // 4) q / k / vc projections (WMMA, LDS-staged B)
  const int M = BB * NT; // 32768
  k_gemm_lds<<<gemm_blocks_lds(M, CC, 1, 1), 256, 0, stream>>>(
      M, CC, CC, 1, 1, xn_h, CC, 0, 0, wq_h, CC, 0, 0, q_b,
      q_f, nullptr, CC, 1, 0, 0, nullptr, nullptr, EPI_F32);
  k_gemm_lds<<<gemm_blocks_lds(M, CC, 1, 1), 256, 0, stream>>>(
      M, CC, CC, 1, 1, xn_h, CC, 0, 0, wk_h, CC, 0, 0, k_b,
      k_f, nullptr, CC, 1, 0, 0, nullptr, nullptr, EPI_F32);
  k_gemm_lds<<<gemm_blocks_lds(M, CC, 1, 1), 256, 0, stream>>>(
      M, CC, CC, 1, 1, xn_h, CC, 0, 0, wv_h, CC, 0, 0, vc_b,
      vc_f, nullptr, CC, 1, 0, 0, nullptr, nullptr, EPI_F32);

  // 5) L2-normalize q,k over tokens -> transposed (B,C,N) f16
  k_l2n<<<BB * CC / 8, 256, 0, stream>>>(q_f, qt_h);
  k_l2n<<<BB * CC / 8, 256, 0, stream>>>(k_f, kt_h);

  // 6) dwconv + BN + SiLU gate -> v
  k_dwgate<<<cdiv(BNC, 256), 256, 0, stream>>>(xn_f, vc_f, dw_w, dw_b, bn_g, bn_b, bn_m, bn_v, v_h);

  // 7) attn = qn @ knT  (per (b,h): 64x64, K=1024)
  k_gemm<<<gemm_blocks(DHh, DHh, BB, NHh), 256, 0, stream>>>(
      DHh, DHh, NT, BB, NHh,
      qt_h, NT, (long long)CC * NT, (long long)DHh * NT,
      kt_h, NT, (long long)CC * NT, (long long)DHh * NT,
      nullptr, attnF, nullptr, DHh, 1, (long long)NHh * DHh * DHh, (long long)DHh * DHh,
      nullptr, nullptr, EPI_F32);

  // 8) temp-scale + softmax
  k_smax<<<BB * NHh * DHh / 8, 256, 0, stream>>>(attnF, temp, attnH);

  // 9) xo = attn @ v  (per (b,h): 64x1024, K=64) -> (B,N,C) f16
  k_gemm<<<gemm_blocks(DHh, NT, BB, NHh), 256, 0, stream>>>(
      DHh, NT, DHh, BB, NHh,
      attnH, DHh, (long long)NHh * DHh * DHh, (long long)DHh * DHh,
      v_h, CC, (long long)NT * CC, DHh,
      nullptr, nullptr, xo_h, 1, CC, (long long)NT * CC, DHh,
      nullptr, nullptr, EPI_F16);

  // 10) proj
  k_gemm_lds<<<gemm_blocks_lds(M, CC, 1, 1), 256, 0, stream>>>(
      M, CC, CC, 1, 1, xo_h, CC, 0, 0, wp_h, CC, 0, 0, proj_b,
      projf, nullptr, CC, 1, 0, 0, nullptr, nullptr, EPI_F32);

  // 11) residual 1 with scramble
  k_resid1<<<cdiv(BNC, 256), 256, 0, stream>>>(x, projf, gamma1, x1);

  // 12) LN2
  k_ln<<<BB * NT / 8, 256, 0, stream>>>(x1, nullptr, ln2_w, ln2_b, nullptr, xn2_h);

  // 13) MLP up + GELU
  k_gemm_lds<<<gemm_blocks_lds(M, HID2, 1, 1), 256, 0, stream>>>(
      M, HID2, CC, 1, 1, xn2_h, CC, 0, 0, w1_h, CC, 0, 0, mlp_b1,
      nullptr, h1_h, HID2, 1, 0, 0, nullptr, nullptr, EPI_GELU);

  // 14) MLP dwconv + GELU
  k_dw2<<<cdiv(BNH, 256), 256, 0, stream>>>(h1_h, mlp_dw, mlp_db, h2_h);

  // 15) MLP down + final residual (writes d_out), batched over B for layout
  k_gemm_lds<<<gemm_blocks_lds(NT, CC, BB, 1), 256, 0, stream>>>(
      NT, CC, HID2, BB, 1,
      h2_h, HID2, (long long)NT * HID2, 0,
      w2_h, HID2, 0, 0, mlp_b2,
      out, nullptr, 1, NT, (long long)CC * NT, 0,
      x1, gamma2, EPI_RES);
}
